// BiLstmCRF_18622978196108
// MI455X (gfx1250) — matrix operand order repaired and checked by reading, not compiled, not run on hardware
//
#include <hip/hip_runtime.h>
#include <hip/hip_bf16.h>

// ---------------- problem constants (from reference) ----------------
#define BQ   16
#define TQ   256
#define DQ   512
#define HQ   256
#define LQ   2
#define NFFQ 2
#define KQ   20
#define NTOK (BQ*TQ)        // 4096
#define G4H  (4*HQ)         // 1024

typedef __bf16 bf16_t;
typedef bf16_t bf16x8  __attribute__((ext_vector_type(8)));
typedef bf16_t bf16x16 __attribute__((ext_vector_type(16)));
typedef float  f32x8   __attribute__((ext_vector_type(8)));
typedef int    v4i_t   __attribute__((ext_vector_type(4)));

typedef __attribute__((address_space(1))) v4i_t* gbl_v4i_p;   // global int4*
typedef __attribute__((address_space(3))) v4i_t* lds_v4i_p;   // LDS int4*

__device__ __forceinline__ f32x8 wmma_bf16(bf16x16 a, bf16x16 b, f32x8 c) {
    // v_wmma_f32_16x16x32_bf16  (neg_a, A, neg_b, B, c_mod, C, reuse_a, reuse_b)
    return __builtin_amdgcn_wmma_f32_16x16x32_bf16(false, a, false, b, (short)0, c, false, false);
}

__device__ __forceinline__ bf16x16 make_a_frag(bf16x8 lo, bf16x8 hi) {
    return __builtin_shufflevector(lo, hi, 0,1,2,3,4,5,6,7,8,9,10,11,12,13,14,15);
}

__device__ __forceinline__ float sigf(float x) { return 1.0f / (1.0f + __expf(-x)); }

// ---- gfx1250 async global->LDS copy (16B per lane), tracked on ASYNCcnt ----
__device__ __forceinline__ void async_copy_b128(const bf16_t* g, bf16_t* l) {
#if defined(__has_builtin) && __has_builtin(__builtin_amdgcn_global_load_async_to_lds_b128)
    __builtin_amdgcn_global_load_async_to_lds_b128(
        (gbl_v4i_p)(void*)g, (lds_v4i_p)(void*)l, 0, 0);
#else
    unsigned int la = (unsigned int)(uintptr_t)(__attribute__((address_space(3))) void*)(void*)l;
    asm volatile("global_load_async_to_lds_b128 %0, %1, off" :: "v"(la), "v"(g) : "memory");
#endif
}

__device__ __forceinline__ void async_wait0() {
#if defined(__has_builtin) && __has_builtin(__builtin_amdgcn_s_wait_asynccnt)
    __builtin_amdgcn_s_wait_asynccnt(0);
#else
    asm volatile("s_wait_asynccnt 0" ::: "memory");
#endif
}

// ---------------- f32 -> bf16 conversion ----------------
__global__ void cvt_bf16_kernel(const float* __restrict__ in, bf16_t* __restrict__ out, int n) {
    int e = blockIdx.x * blockDim.x + threadIdx.x;
    if (e < n) out[e] = (bf16_t)in[e];
}

// ---------------- embedding gather -> bf16 ----------------
__global__ void embed_bf16_kernel(const int* __restrict__ x, const float* __restrict__ emb,
                                  bf16_t* __restrict__ out) {
    int e = blockIdx.x * blockDim.x + threadIdx.x;
    if (e >= NTOK * DQ) return;
    int tok = e / DQ, d = e % DQ;
    out[e] = (bf16_t)emb[(size_t)x[tok] * DQ + d];
}

// ---------------- WMMA GEMM: C = A(MxK) * W(NxK)^T + bias ----------------
// 64x128 macro-tile per block (256 threads = 8 waves). The 64-row A strip is
// staged to LDS with GLOBAL_LOAD_ASYNC_TO_LDS_B128 and shared by all 8 waves;
// each wave's 32B B fragment is reused across 4 WMMAs (4 m-tiles).
// Requires Kd <= 512, M % 64 == 0, N % 128 == 0.
__global__ __launch_bounds__(256, 2)
void gemm_bf16_wmma(const bf16_t* __restrict__ A, const bf16_t* __restrict__ W,
                    const float* __restrict__ bias, const float* __restrict__ resid,
                    float* __restrict__ Cf, bf16_t* __restrict__ Cb,
                    int M, int N, int Kd, int relu) {
    __shared__ bf16_t a_sh[64 * DQ];                 // 64 KB A strip
    const int tid  = threadIdx.x;
    const int lane = tid & 31;
    const int wave = tid >> 5;
    const int hlf  = lane >> 4;                      // 0: lanes 0-15, 1: lanes 16-31
    const int lcol = lane & 15;
    const int m0   = blockIdx.y * 64;
    const int n0   = (blockIdx.x * 8 + wave) * 16;

    // --- async stage A[m0..m0+64) x Kd -> LDS (uniform trip count: EXEC all-ones) ---
    const int kchunks = Kd >> 3;                     // 16B chunks per row
    for (int idx = tid; idx < 64 * kchunks; idx += 256) {
        const int row = idx / kchunks;
        const int ko  = (idx - row * kchunks) << 3;
        async_copy_b128(A + (size_t)(m0 + row) * Kd + ko, a_sh + row * Kd + ko);
    }
    async_wait0();
    __syncthreads();

    const bf16_t* Wr = W + (size_t)(n0 + lcol) * Kd + hlf * 16;  // B-frag: row of W
    const bf16_t* As = a_sh + (size_t)lcol * Kd + hlf * 16;      // A-frag base (mt=0)

    f32x8 acc[4] = {};
    for (int kk = 0; kk < Kd; kk += 32) {
        bf16x16 bfr = *(const bf16x16*)(Wr + kk);
        if (kk + 32 < Kd) __builtin_prefetch(Wr + kk + 32, 0, 1);  // global_prefetch_b8
        #pragma unroll
        for (int mt = 0; mt < 4; ++mt) {
            const bf16_t* ap = As + (size_t)(mt * 16) * Kd + kk;
            bf16x16 afrag = make_a_frag(*(const bf16x8*)ap, *(const bf16x8*)(ap + 8));
            acc[mt] = wmma_bf16(afrag, bfr, acc[mt]);
        }
    }

    const int col = n0 + lcol;
    const float bv = bias ? bias[col] : 0.0f;
    #pragma unroll
    for (int mt = 0; mt < 4; ++mt) {
        #pragma unroll
        for (int r = 0; r < 8; ++r) {
            const int row = m0 + mt * 16 + r + hlf * 8;  // C layout: VGPR r -> M=r (+8 hi lanes)
            float v = acc[mt][r] + bv;
            if (relu) v = fmaxf(v, 0.0f);
            if (resid) v += resid[(size_t)row * N + col];
            if (Cf) Cf[(size_t)row * N + col] = v;
            if (Cb) Cb[(size_t)row * N + col] = (bf16_t)v;
        }
    }
}

// ---------------- persistent BiLSTM layer kernel ----------------
// grid = 2 blocks (blockIdx.x = direction), 512 threads = 16 waves.
// Wave w owns gate tiles {w, w+16, w+32, w+48} of g = h @ Whh^T, i.e. all four
// gates (i,f,g,o) for hidden slice [16w, 16w+16) -> gate math stays in registers.
// h lives in LDS as bf16 (A operand of the per-step WMMA); cell state c in VGPRs.
__global__ __launch_bounds__(512, 1)
void lstm_layer_kernel(const float* __restrict__ xw_f, const float* __restrict__ xw_b,
                       const bf16_t* __restrict__ WhhF, const bf16_t* __restrict__ WhhB,
                       float* __restrict__ outF, bf16_t* __restrict__ outB) {
    __shared__ bf16_t h_sh[BQ * HQ];              // 8 KB

    const int dir = blockIdx.x;
    const float*  xw  = dir ? xw_b : xw_f;
    const bf16_t* Whh = dir ? WhhB : WhhF;
    const int colbase = dir ? HQ : 0;

    const int lane = threadIdx.x & 31;
    const int wave = threadIdx.x >> 5;            // 0..15
    const int hlf  = lane >> 4;
    const int lcol = lane & 15;
    const int hh   = wave * 16 + lcol;            // hidden index 0..255

    for (int e = threadIdx.x; e < BQ * HQ; e += blockDim.x) h_sh[e] = (bf16_t)0.0f;

    // B-frag base pointers for the 4 gate tiles this wave owns
    const bf16_t* Wi = Whh + (size_t)((wave     ) * 16 + lcol) * HQ + hlf * 16;
    const bf16_t* Wf = Whh + (size_t)((wave + 16) * 16 + lcol) * HQ + hlf * 16;
    const bf16_t* Wg = Whh + (size_t)((wave + 32) * 16 + lcol) * HQ + hlf * 16;
    const bf16_t* Wo = Whh + (size_t)((wave + 48) * 16 + lcol) * HQ + hlf * 16;
    const bf16_t* Ash = h_sh + (size_t)lcol * HQ + hlf * 16;

    f32x8 c = {};                                  // cell state, accumulator layout
    __syncthreads();

    for (int t = 0; t < TQ; ++t) {
        const int tt = dir ? (TQ - 1 - t) : t;

        f32x8 ai = {}, af_ = {}, ag = {}, ao = {};
        #pragma unroll
        for (int kk = 0; kk < HQ; kk += 32) {
            bf16x8  a0 = *(const bf16x8*)(Ash + kk);
            bf16x8  a1 = *(const bf16x8*)(Ash + kk + 8);
            bf16x16 afrag = make_a_frag(a0, a1);
            bf16x16 bi  = *(const bf16x16*)(Wi + kk);
            bf16x16 bff = *(const bf16x16*)(Wf + kk);
            bf16x16 bg  = *(const bf16x16*)(Wg + kk);
            bf16x16 bo  = *(const bf16x16*)(Wo + kk);
            ai  = wmma_bf16(afrag, bi,  ai);
            af_ = wmma_bf16(afrag, bff, af_);
            ag  = wmma_bf16(afrag, bg,  ag);
            ao  = wmma_bf16(afrag, bo,  ao);
        }
        __syncthreads();                           // all h_sh reads done

        #pragma unroll
        for (int r = 0; r < 8; ++r) {
            const int row = r + hlf * 8;           // batch index 0..15
            const float* xwr = xw + ((size_t)row * TQ + tt) * G4H;
            float gi = ai[r]  + xwr[hh];
            float gf = af_[r] + xwr[HQ + hh];
            float gg = ag[r]  + xwr[2 * HQ + hh];
            float go = ao[r]  + xwr[3 * HQ + hh];
            float cv = sigf(gf) * c[r] + sigf(gi) * tanhf(gg);
            float hv = sigf(go) * tanhf(cv);
            c[r] = cv;
            const size_t o = ((size_t)row * TQ + tt) * (2 * HQ) + colbase + hh;
            outF[o] = hv;
            outB[o] = (bf16_t)hv;
            h_sh[row * HQ + hh] = (bf16_t)hv;
        }
        __syncthreads();                           // h_sh ready for next step
    }
}

// ---------------- emissions: em = h(4096x256) @ out_W(20x256)^T + out_b ----------------
__global__ void em_kernel(const float* __restrict__ h, const float* __restrict__ W,
                          const float* __restrict__ b, float* __restrict__ em) {
    int gid = blockIdx.x * blockDim.x + threadIdx.x;
    if (gid >= NTOK * KQ) return;
    int tok = gid / KQ, k = gid % KQ;
    const float* hr = h + (size_t)tok * HQ;
    const float* wr = W + (size_t)k * HQ;
    float s = b[k];
    for (int j = 0; j < HQ; ++j) s += hr[j] * wr[j];
    em[gid] = s;
}

// ---------------- Viterbi: one wave per batch, lane = state ----------------
__global__ void viterbi_kernel(const float* __restrict__ em, const float* __restrict__ startv,
                               const float* __restrict__ endv, const float* __restrict__ trans,
                               int* __restrict__ hist, float* __restrict__ tag_out) {
    const int b = blockIdx.x;
    const int lane = threadIdx.x;                  // 0..31, states 0..19 valid
    __shared__ int tags[TQ];
    const float NEG = -3.0e38f;

    float score = (lane < KQ) ? startv[lane] + em[((size_t)b * TQ) * KQ + lane] : NEG;
    for (int t = 1; t < TQ; ++t) {
        float best = NEG; int arg = 0;
        for (int i = 0; i < KQ; ++i) {
            float si = __shfl(score, i, 32);
            float v = si + ((lane < KQ) ? trans[i * KQ + lane] : 0.0f);
            if (v > best) { best = v; arg = i; }
        }
        if (lane < KQ) hist[((size_t)(t - 1) * BQ + b) * KQ + lane] = arg;
        score = (lane < KQ) ? best + em[((size_t)b * TQ + t) * KQ + lane] : NEG;
    }
    float fin = (lane < KQ) ? score + endv[lane] : NEG;
    float bmax = NEG; int btag = 0;
    for (int i = 0; i < KQ; ++i) {
        float si = __shfl(fin, i, 32);
        if (si > bmax) { bmax = si; btag = i; }
    }
    if (lane == 0) {
        int tg = btag;
        tags[TQ - 1] = tg;
        for (int t = TQ - 2; t >= 0; --t) {
            tg = hist[((size_t)t * BQ + b) * KQ + tg];
            tags[t] = tg;
        }
    }
    __syncthreads();
    for (int t = lane; t < TQ; t += 32) tag_out[(size_t)b * TQ + t] = (float)tags[t];
}

// ---------------- host orchestration ----------------
static inline void cvt(const float* in, bf16_t* out, int n, hipStream_t s) {
    cvt_bf16_kernel<<<(n + 255) / 256, 256, 0, s>>>(in, out, n);
}

extern "C" void kernel_launch(void* const* d_in, const int* in_sizes, int n_in,
                              void* d_out, int out_size, void* d_ws, size_t ws_size,
                              hipStream_t stream) {
    const int*   x        = (const int*)  d_in[0];
    const float* emb      = (const float*)d_in[1];
    const float* lstm_Wih = (const float*)d_in[2];   // (2,2,1024,512)
    const float* lstm_Whh = (const float*)d_in[3];   // (2,2,1024,256)
    const float* lstm_b   = (const float*)d_in[4];   // (2,2,1024)
    const float* ff_W1    = (const float*)d_in[5];   // (2,512,512)
    const float* ff_b1    = (const float*)d_in[6];
    const float* ff_W2    = (const float*)d_in[7];
    const float* ff_b2    = (const float*)d_in[8];
    const float* ffo_W    = (const float*)d_in[9];   // (256,512)
    const float* ffo_b    = (const float*)d_in[10];
    const float* out_W    = (const float*)d_in[11];  // (20,256)
    const float* out_b    = (const float*)d_in[12];
    const float* crf_s    = (const float*)d_in[13];
    const float* crf_e    = (const float*)d_in[14];
    const float* crf_t    = (const float*)d_in[15];

    float* em_out  = (float*)d_out;                  // (B,T,K)
    float* tag_out = em_out + (size_t)NTOK * KQ;     // (B,T) as float

    // --- workspace bump allocator ---
    char* wp = (char*)d_ws;
    auto alloc = [&](size_t bytes) -> void* {
        void* p = (void*)wp;
        wp += (bytes + 255) & ~(size_t)255;
        return p;
    };
    bf16_t* wih_bf = (bf16_t*)alloc((size_t)LQ*2*G4H*DQ * 2);
    bf16_t* whh_bf = (bf16_t*)alloc((size_t)LQ*2*G4H*HQ * 2);
    bf16_t* fw1_bf = (bf16_t*)alloc((size_t)NFFQ*DQ*DQ * 2);
    bf16_t* fw2_bf = (bf16_t*)alloc((size_t)NFFQ*DQ*DQ * 2);
    bf16_t* fwo_bf = (bf16_t*)alloc((size_t)HQ*DQ * 2);
    bf16_t* emb_bf = (bf16_t*)alloc((size_t)NTOK*DQ * 2);
    float*  xw0    = (float*) alloc((size_t)NTOK*G4H * 4);
    float*  xw1    = (float*) alloc((size_t)NTOK*G4H * 4);
    float*  l1_f   = (float*) alloc((size_t)NTOK*DQ * 4);
    bf16_t* l1_b   = (bf16_t*)alloc((size_t)NTOK*DQ * 2);
    float*  l2_f   = (float*) alloc((size_t)NTOK*DQ * 4);
    bf16_t* l2_b   = (bf16_t*)alloc((size_t)NTOK*DQ * 2);
    float*  fft_f  = (float*) alloc((size_t)NTOK*DQ * 4);
    bf16_t* fft_b  = (bf16_t*)alloc((size_t)NTOK*DQ * 2);
    float*  ffc_f  = (float*) alloc((size_t)NTOK*DQ * 4);
    bf16_t* ffc_b  = (bf16_t*)alloc((size_t)NTOK*DQ * 2);
    float*  ffo_o  = (float*) alloc((size_t)NTOK*HQ * 4);
    int*    hist   = (int*)   alloc((size_t)(TQ-1)*BQ*KQ * 4);
    (void)n_in; (void)in_sizes; (void)out_size; (void)ws_size;

    // --- weight conversions (fp32 -> bf16) ---
    cvt(lstm_Wih, wih_bf, LQ*2*G4H*DQ, stream);
    cvt(lstm_Whh, whh_bf, LQ*2*G4H*HQ, stream);
    cvt(ff_W1,    fw1_bf, NFFQ*DQ*DQ,  stream);
    cvt(ff_W2,    fw2_bf, NFFQ*DQ*DQ,  stream);
    cvt(ffo_W,    fwo_bf, HQ*DQ,       stream);

    // --- embedding gather ---
    embed_bf16_kernel<<<(NTOK*DQ + 255)/256, 256, 0, stream>>>(x, emb, emb_bf);

    const dim3 blk(256);
    // --- BiLSTM layers ---
    const bf16_t* lin = emb_bf;
    for (int l = 0; l < LQ; ++l) {
        const bf16_t* WihF = wih_bf + (size_t)(l*2 + 0) * G4H * DQ;
        const bf16_t* WihB = wih_bf + (size_t)(l*2 + 1) * G4H * DQ;
        const bf16_t* WhhF = whh_bf + (size_t)(l*2 + 0) * G4H * HQ;
        const bf16_t* WhhB = whh_bf + (size_t)(l*2 + 1) * G4H * HQ;
        const float*  bF   = lstm_b + (size_t)(l*2 + 0) * G4H;
        const float*  bB   = lstm_b + (size_t)(l*2 + 1) * G4H;
        // xw = x @ Wih^T + b   (M=4096, N=1024, K=512)
        gemm_bf16_wmma<<<dim3(G4H/128, NTOK/64), blk, 0, stream>>>(
            lin, WihF, bF, nullptr, xw0, nullptr, NTOK, G4H, DQ, 0);
        gemm_bf16_wmma<<<dim3(G4H/128, NTOK/64), blk, 0, stream>>>(
            lin, WihB, bB, nullptr, xw1, nullptr, NTOK, G4H, DQ, 0);
        float*  lof = (l == 0) ? l1_f : l2_f;
        bf16_t* lob = (l == 0) ? l1_b : l2_b;
        lstm_layer_kernel<<<2, 512, 0, stream>>>(xw0, xw1, WhhF, WhhB, lof, lob);
        lin = lob;
    }

    // --- FF blocks: h = relu(h@W1^T+b1)@W2^T+b2  (x2), residual after loop ---
    // iter 0
    gemm_bf16_wmma<<<dim3(DQ/128, NTOK/64), blk, 0, stream>>>(
        l2_b, fw1_bf, ff_b1, nullptr, fft_f, fft_b, NTOK, DQ, DQ, 1);
    gemm_bf16_wmma<<<dim3(DQ/128, NTOK/64), blk, 0, stream>>>(
        fft_b, fw2_bf, ff_b2, nullptr, ffc_f, ffc_b, NTOK, DQ, DQ, 0);
    // iter 1 (+ residual of lstm output, fused into epilogue)
    gemm_bf16_wmma<<<dim3(DQ/128, NTOK/64), blk, 0, stream>>>(
        ffc_b, fw1_bf + (size_t)DQ*DQ, ff_b1 + DQ, nullptr, fft_f, fft_b, NTOK, DQ, DQ, 1);
    gemm_bf16_wmma<<<dim3(DQ/128, NTOK/64), blk, 0, stream>>>(
        fft_b, fw2_bf + (size_t)DQ*DQ, ff_b2 + DQ, l2_f, l1_f, l1_b, NTOK, DQ, DQ, 0);

    // --- ffo: (4096x512) -> (4096x256) ---
    gemm_bf16_wmma<<<dim3(HQ/128, NTOK/64), blk, 0, stream>>>(
        l1_b, fwo_bf, ffo_b, nullptr, ffo_o, nullptr, NTOK, HQ, DQ, 0);

    // --- emissions (N=20, tiny: plain f32 dot kernel) ---
    em_kernel<<<(NTOK*KQ + 255)/256, 256, 0, stream>>>(ffo_o, out_W, out_b, em_out);

    // --- Viterbi decode ---
    viterbi_kernel<<<BQ, 32, 0, stream>>>(em_out, crf_s, crf_e, crf_t, hist, tag_out);
}
